// RVQQuantizer_34488587387009
// MI455X (gfx1250) — matrix operand order repaired and checked by reading, not compiled, not run on hardware
//
#include <hip/hip_runtime.h>
#include <hip/hip_bf16.h>

typedef __attribute__((ext_vector_type(16))) __bf16        v16bf;
typedef __attribute__((ext_vector_type(8)))  float         v8f;
typedef __attribute__((ext_vector_type(8)))  unsigned int  v8u;

#define DIM       256
#define KCODES    4096
#define NTOK      16384
#define NROWTILES (NTOK / 16)    // 1024
#define NCOLTILES (KCODES / 16)  // 256
#define LN2F      0.69314718055994530942f

__device__ __forceinline__ unsigned short f2bf_rne(float f) {
    unsigned int x = __float_as_uint(f);
    unsigned int r = x + 0x7FFFu + ((x >> 16) & 1u);
    return (unsigned short)(r >> 16);
}

// ---------------------------------------------------------------------------
// Kernel 1: L2-normalize rows of [nrows,256] fp32, convert to bf16 (RNE), and
// write into WMMA operand-swizzled layout:
//   dst element layout: [tile][kc(0..7)][lane(0..31)][16 bf16]  (32B per lane)
// mode 0 = A-matrix layout (16x32 bf16 tile, ISA 7.12.2)
// mode 1 = B-matrix layout (32x16 bf16 tile)
// ---------------------------------------------------------------------------
__global__ __launch_bounds__(512) void vq_norm_swizzle(
    const float* __restrict__ src, unsigned short* __restrict__ dst, int mode) {
    __shared__ unsigned short tile[16][DIM];  // 8 KB
    const int t   = blockIdx.x;
    const int tid = threadIdx.x;
    const int w   = tid >> 5;   // wave = row within tile
    const int l   = tid & 31;

    const float* r = src + (size_t)(t * 16 + w) * DIM;
    float vals[8];
    float ss = 0.f;
#pragma unroll
    for (int j = 0; j < 8; ++j) {
        float v = r[l * 8 + j];
        vals[j] = v;
        ss += v * v;
    }
#pragma unroll
    for (int off = 16; off >= 1; off >>= 1) ss += __shfl_xor(ss, off, 32);
    const float inv = 1.0f / fmaxf(sqrtf(ss), 1e-12f);
#pragma unroll
    for (int j = 0; j < 8; ++j) tile[w][l * 8 + j] = f2bf_rne(vals[j] * inv);
    __syncthreads();

    if (tid < 256) {
        const int kc = tid >> 5, lane = tid & 31;
        const int rloc = lane & 15, half = lane >> 4;
        unsigned short out[16];
#pragma unroll
        for (int i = 0; i < 16; ++i) {
            int kloc;
            if (mode == 0)  // A: VGPR v holds K pairs split 0..7 / 16..23 by half
                kloc = kc * 32 + ((i < 8) ? (half * 8 + i) : (16 + half * 8 + (i - 8)));
            else            // B: same K split as A (0..7 / 16..23 by half)
                kloc = kc * 32 + ((i < 8) ? (half * 8 + i) : (16 + half * 8 + (i - 8)));
            out[i] = tile[rloc][kloc];
        }
        v8u pack;
#pragma unroll
        for (int j = 0; j < 8; ++j)
            pack[j] = (unsigned int)out[2 * j] | ((unsigned int)out[2 * j + 1] << 16);
        ((v8u*)dst)[(size_t)(t * 8 + kc) * 32 + lane] = pack;
    }
}

// ---------------------------------------------------------------------------
// Kernel 2: fused  cos_sim GEMM (WMMA bf16) + gumbel + argmax + gather +
// commitment partial.  One block (8 waves) per 16-token row tile.
// ---------------------------------------------------------------------------
__global__ __launch_bounds__(256) void vq_assign(
    const unsigned short* __restrict__ zA, const unsigned short* __restrict__ cbB,
    const float* __restrict__ u, const float* __restrict__ z,
    const float* __restrict__ codebook, float* __restrict__ zq,
    float* __restrict__ emb, float* __restrict__ idx_out,
    float* __restrict__ partials) {
    __shared__ float sMax[8][16];
    __shared__ int   sIdx[8][16];
    __shared__ int   sFinal[16];
    __shared__ float sCommit[8];

    const int rt   = blockIdx.x;
    const int tid  = threadIdx.x;
    const int w    = tid >> 5;
    const int lane = tid & 31;
    const int half = lane >> 4;
    const int col  = lane & 15;

    // A tile (16 tokens x 256 dims) stays resident in 64 VGPRs
    const v8u* zA8 = (const v8u*)zA;
    v16bf a[8];
#pragma unroll
    for (int kc = 0; kc < 8; ++kc)
        a[kc] = __builtin_bit_cast(v16bf, zA8[(size_t)(rt * 8 + kc) * 32 + lane]);

    float bestv[8];
    int   besti[8];
#pragma unroll
    for (int v = 0; v < 8; ++v) { bestv[v] = -3.0e38f; besti[v] = 0; }

    const v8u* cb8 = (const v8u*)cbB;
    for (int it = 0; it < NCOLTILES / 8; ++it) {
        const int ct = w + it * 8;  // column tile for this wave
        v8f acc = {};
#pragma unroll
        for (int kc = 0; kc < 8; ++kc) {
            v16bf b = __builtin_bit_cast(v16bf, cb8[(size_t)(ct * 8 + kc) * 32 + lane]);
            acc = __builtin_amdgcn_wmma_f32_16x16x32_bf16(
                false, a[kc], false, b, (short)0, acc, false, false);
        }
        const int code = ct * 16 + col;
#pragma unroll
        for (int v = 0; v < 8; ++v) {
            const int token = rt * 16 + v + half * 8;  // C layout: M = v + 8*half
            float uu = __builtin_nontemporal_load(&u[(size_t)token * KCODES + code]);
            // gumbel in log2 form; args are never denormal so a raw v_log_f32
            // (base-2, no OCML denorm guards) is exact enough:
            //   t = -ln(u+eps)+eps ; score = cos - ln(t) = cos + gumbel
            float l2i = __builtin_amdgcn_logf(uu + 1e-10f);
            float tt  = __builtin_fmaf(-LN2F, l2i, 1e-10f);
            float l2o = __builtin_amdgcn_logf(tt);
            // (logits+gumbel)/tau is monotone in cos+g; per-token constants
            // (-1, tau) cannot change the argmax.
            float score = __builtin_fmaf(-LN2F, l2o, acc[v]);
            if (score > bestv[v]) { bestv[v] = score; besti[v] = code; }
        }
    }

    // argmax across the 16 lanes of each half (rows differ between halves)
#pragma unroll
    for (int v = 0; v < 8; ++v) {
        float bv = bestv[v];
        int   bi = besti[v];
#pragma unroll
        for (int off = 8; off >= 1; off >>= 1) {
            float ov = __shfl_xor(bv, off, 32);
            int   oi = __shfl_xor(bi, off, 32);
            if (ov > bv || (ov == bv && oi < bi)) { bv = ov; bi = oi; }
        }
        bestv[v] = bv;
        besti[v] = bi;
    }
    if (col == 0) {
#pragma unroll
        for (int v = 0; v < 8; ++v) {
            sMax[w][v + half * 8] = bestv[v];
            sIdx[w][v + half * 8] = besti[v];
        }
    }
    __syncthreads();

    // argmax across the 8 waves
    if (tid < 16) {
        float bv = sMax[0][tid];
        int   bi = sIdx[0][tid];
#pragma unroll
        for (int ww = 1; ww < 8; ++ww) {
            float ov = sMax[ww][tid];
            int   oi = sIdx[ww][tid];
            if (ov > bv || (ov == bv && oi < bi)) { bv = ov; bi = oi; }
        }
        sFinal[tid] = bi;
        idx_out[rt * 16 + tid] = (float)bi;
    }
    __syncthreads();

    // gather codebook rows -> z_q and embedding; commitment partial
    const int r     = tid >> 4;
    const int c0    = tid & 15;  // lanes 0-15 cover contiguous 64B per j
    const int token = rt * 16 + r;
    const int ci    = sFinal[r];
    const float* crow = codebook + (size_t)ci * DIM;
    const float* zrow = z + (size_t)token * DIM;
    float* q1 = zq + (size_t)token * DIM;
    float* q2 = emb + (size_t)token * DIM;
    float local = 0.f;
#pragma unroll
    for (int j = 0; j < 16; ++j) {
        const int d = c0 + j * 16;
        float c = crow[d];
        float zv = zrow[d];
        q1[d] = c;
        q2[d] = c;
        float diff = c - zv;
        local += diff * diff;
    }
#pragma unroll
    for (int off = 16; off >= 1; off >>= 1) local += __shfl_xor(local, off, 32);
    if (lane == 0) sCommit[w] = local;
    __syncthreads();
    if (tid == 0) {
        float s = 0.f;
#pragma unroll
        for (int ww = 0; ww < 8; ++ww) s += sCommit[ww];
        partials[rt] = s;  // deterministic: fixed order, no atomics
    }
}

// ---------------------------------------------------------------------------
// Kernel 3: scalars (commitment, entropy bonus, perplexity, entropy)
// ---------------------------------------------------------------------------
__global__ __launch_bounds__(256) void vq_finalize(
    const float* __restrict__ usage, const int* __restrict__ step_ptr,
    const float* __restrict__ partials, float* __restrict__ outs) {
    __shared__ float red[256];
    const int tid = threadIdx.x;

    float cs = 0.f;
    for (int i = tid; i < NROWTILES; i += 256) cs += partials[i];
    red[tid] = cs;
    __syncthreads();
    for (int off = 128; off >= 1; off >>= 1) {
        if (tid < off) red[tid] += red[tid + off];
        __syncthreads();
    }
    const float commit = 0.5f * red[0] / (float)((size_t)NTOK * DIM);
    __syncthreads();

    float us = 0.f;
    for (int i = tid; i < KCODES; i += 256) us += usage[i];
    red[tid] = us;
    __syncthreads();
    for (int off = 128; off >= 1; off >>= 1) {
        if (tid < off) red[tid] += red[tid + off];
        __syncthreads();
    }
    const float usum = red[0];
    __syncthreads();

    float ep = 0.f;
    for (int i = tid; i < KCODES; i += 256) {
        float p = (usum > 0.f) ? usage[i] / (usum + 1e-10f) : (1.0f / (float)KCODES);
        ep += -p * logf(p + 1e-10f);  // accuracy matters for entropy output
    }
    red[tid] = ep;
    __syncthreads();
    for (int off = 128; off >= 1; off >>= 1) {
        if (tid < off) red[tid] += red[tid + off];
        __syncthreads();
    }

    if (tid == 0) {
        float entropy = red[0];
        float perp    = expf(entropy);
        int step_after = step_ptr[0] + 1;
        float bw    = 0.05f * (1.0f - (float)step_after / 20000.0f);
        float bonus = (step_after < 20000) ? (-bw * entropy) : 0.0f;
        outs[0] = commit;
        outs[1] = bonus;
        outs[2] = perp;
        outs[3] = entropy;
    }
}

extern "C" void kernel_launch(void* const* d_in, const int* in_sizes, int n_in,
                              void* d_out, int out_size, void* d_ws, size_t ws_size,
                              hipStream_t stream) {
    (void)in_sizes; (void)n_in; (void)out_size; (void)ws_size;
    const float* z     = (const float*)d_in[0];
    const float* u     = (const float*)d_in[1];
    const float* cb    = (const float*)d_in[2];
    const float* usage = (const float*)d_in[3];
    const int*   step  = (const int*)d_in[4];

    unsigned short* zA  = (unsigned short*)d_ws;                 // 8 MB
    unsigned short* cbB = zA + (size_t)NTOK * DIM;               // 2 MB
    float* partials     = (float*)(cbB + (size_t)KCODES * DIM);  // 4 KB

    float* zq      = (float*)d_out;
    float* emb     = zq + (size_t)NTOK * DIM;
    float* idx_out = emb + (size_t)NTOK * DIM;
    float* scalars = idx_out + NTOK;

    vq_norm_swizzle<<<NROWTILES, 512, 0, stream>>>(z, zA, 0);
    vq_norm_swizzle<<<KCODES / 16, 512, 0, stream>>>(cb, cbB, 1);
    vq_assign<<<NROWTILES, 256, 0, stream>>>(zA, cbB, u, z, cb, zq, emb, idx_out, partials);
    vq_finalize<<<1, 256, 0, stream>>>(usage, step, partials, scalars);
}
